// MemoryBank_24240795419545
// MI455X (gfx1250) — compile-verified, matrix operand order checked by reading, not run on hardware
//
#include <hip/hip_runtime.h>
#include <hip/hip_bf16.h>
#include <stdint.h>

// MI455X fused memory-bank attention:
//   S = Q @ mem^T ; P = softmax(S) ; O = P @ mem
// bf16 WMMA (v_wmma_f32_16x16x32_bf16) with fp32 accumulate; fp32 softmax.
// 16 wave32 per workgroup (4 waves/SIMD), full score row in 256 KB LDS.

typedef __attribute__((ext_vector_type(16))) __bf16 v16bf;
typedef __attribute__((ext_vector_type(8)))  float  v8f;

union V16 {
  v16bf v;
  uint4 q[2];   // two 16-byte chunks (8 bf16 each)
};

#define BT      32768   // B*T
#define DK      128     // D
#define MK      4096    // M
#define ROWS    16      // query rows per workgroup
#define THREADS 512     // 16 wave32
#define WAVES   16

// ---------------------------------------------------------------------------
// Kernel 0: memory fp32 -> bf16, both row-major [M][D] and transposed [D][M].
// ---------------------------------------------------------------------------
__global__ void cvt_mem_kernel(const float* __restrict__ mem,
                               __bf16* __restrict__ memB,
                               __bf16* __restrict__ memT) {
  int idx = blockIdx.x * blockDim.x + threadIdx.x;   // over M*D
  int m = idx >> 7;
  int d = idx & (DK - 1);
  __bf16 h = (__bf16)mem[idx];
  memB[idx] = h;
  memT[(size_t)d * MK + m] = h;
}

// ---------------------------------------------------------------------------
// Kernel 1: fused GEMM1 + softmax + GEMM2 for a 16-row band.
// ---------------------------------------------------------------------------
__global__ __launch_bounds__(THREADS, 1)
void fused_attn_kernel(const float* __restrict__ Q,       // [BT][D] fp32
                       const __bf16* __restrict__ memB,   // [M][D]  bf16
                       const __bf16* __restrict__ memT,   // [D][M]  bf16
                       float* __restrict__ outF,          // [BT][D] fp32
                       float* __restrict__ outA) {        // [BT][M] fp32
  __shared__ float  s_lds[ROWS * MK];        // 256 KB: scores -> probs
  __shared__ __bf16 q_lds[ROWS * DK];        // 4 KB
  __shared__ float  red_lds[8 * 256];        // 8 KB: GEMM2 k-split partials

  const int tid  = threadIdx.x;
  const int lane = tid & 31;
  const int wave = tid >> 5;
  const int ln   = lane & 15;   // row (A) / column (B,C) index
  const int hi   = lane >> 4;   // lane-half selects K sub-chunk
  const int row0 = blockIdx.x * ROWS;

  // ---- stage Q tile as bf16 in LDS (coalesced) ----
  for (int i = tid; i < ROWS * DK; i += THREADS)
    q_lds[i] = (__bf16)Q[(size_t)row0 * DK + i];
  __syncthreads();

  // ---- hoist the 4 A-operands (Q rows reused across all m-tiles) ----
  // A 16x32 bf16 layout: lane<16 -> K{0..7,16..23}, lane>=16 -> K{8..15,24..31}
  V16 a[4];
#pragma unroll
  for (int kk = 0; kk < 4; ++kk) {
    const __bf16* p = &q_lds[ln * DK + kk * 32 + 8 * hi];
    a[kk].q[0] = *(const uint4*)(p);
    a[kk].q[1] = *(const uint4*)(p + 16);
  }

  // ---- GEMM1: S[16 x 4096] = Q @ mem^T ----------------------------------
  // Each wave covers a contiguous slab of 16 m-tiles; B tiles double-buffered
  // in registers so next-tile global loads overlap the current wmma chain.
  {
    const int t0 = wave * 16;
    V16 b[2][4];

    auto load_b = [&](int t, V16 (&bb)[4]) {
      const __bf16* base = &memB[(size_t)(t * 16 + ln) * DK + 8 * hi];
#pragma unroll
      for (int kk = 0; kk < 4; ++kk) {
        bb[kk].q[0] = *(const uint4*)(base + kk * 32);
        bb[kk].q[1] = *(const uint4*)(base + kk * 32 + 16);
      }
    };

    load_b(t0, b[0]);
    for (int i = 0; i < 16; ++i) {
      if (i + 2 < 16)   // pull tile-after-next toward near cache (L2 -> WGP$)
        __builtin_prefetch(&memB[(size_t)((t0 + i + 2) * 16 + ln) * DK], 0, 3);
      if (i + 1 < 16)
        load_b(t0 + i + 1, b[(i + 1) & 1]);

      V16 (&bc)[4] = b[i & 1];
      v8f acc = {};
#pragma unroll
      for (int kk = 0; kk < 4; ++kk)
        acc = __builtin_amdgcn_wmma_f32_16x16x32_bf16(
            false, a[kk].v, false, bc[kk].v, (short)0, acc, false, false);

      // C layout: lane = n, vgpr v = row (m = v + 8*hi)
      const int m0 = (t0 + i) * 16;
#pragma unroll
      for (int v = 0; v < 8; ++v)
        s_lds[(v + 8 * hi) * MK + m0 + ln] = acc[v];
    }
  }
  __syncthreads();

  // ---- softmax: one wave per row, full-wave shfl_xor reductions ----------
  {
    const int row = wave;                 // 16 waves == 16 rows
    float* srow = &s_lds[row * MK];

    float mx = -3.402823e38f;
    for (int m = lane; m < MK; m += 32) mx = fmaxf(mx, srow[m]);
#pragma unroll
    for (int o = 16; o >= 1; o >>= 1) mx = fmaxf(mx, __shfl_xor(mx, o));

    float sum = 0.f;
    for (int m = lane; m < MK; m += 32) {
      float p = __expf(srow[m] - mx);
      srow[m] = p;
      sum += p;
    }
#pragma unroll
    for (int o = 16; o >= 1; o >>= 1) sum += __shfl_xor(sum, o);
    const float inv = 1.f / sum;

    float* arow = outA + (size_t)(row0 + row) * MK;
    for (int m = lane; m < MK; m += 32) {
      float p = srow[m] * inv;
      srow[m] = p;                               // keep normalized for GEMM2
      __builtin_nontemporal_store(p, arow + m);  // 512 MB stream: bypass cache
    }
  }
  __syncthreads();

  // ---- GEMM2: O[16 x 128] = P @ mem --------------------------------------
  // n-tile = wave & 7 ; k-range split across wave pairs, LDS reduction.
  {
    const int n0    = (wave & 7) * 16;
    const int khalf = wave >> 3;
    const int kbeg  = khalf * (MK / 2);
    const int kend  = kbeg + (MK / 2);

    v8f acc = {};
    for (int k0 = kbeg; k0 < kend; k0 += 32) {
      // A operand: normalized probs fp32 (LDS) -> bf16
      V16 a2;
      const float* apA = &s_lds[ln * MK + k0 + 8 * hi];
      const float* apB = apA + 16;
#pragma unroll
      for (int e = 0; e < 8; ++e) a2.v[e]     = (__bf16)apA[e];
#pragma unroll
      for (int e = 0; e < 8; ++e) a2.v[e + 8] = (__bf16)apB[e];

      // B operand: mem column slice from transposed bf16 copy (contiguous)
      V16 b2;
      const __bf16* bp = &memT[(size_t)(n0 + ln) * MK + k0 + 8 * hi];
      b2.q[0] = *(const uint4*)(bp);
      b2.q[1] = *(const uint4*)(bp + 16);

      acc = __builtin_amdgcn_wmma_f32_16x16x32_bf16(
          false, a2.v, false, b2.v, (short)0, acc, false, false);
    }

    // upper-half waves park partials in LDS
    if (khalf == 1) {
#pragma unroll
      for (int v = 0; v < 8; ++v)
        red_lds[(wave & 7) * 256 + v * 32 + lane] = acc[v];
    }
    __syncthreads();

    if (khalf == 0) {
#pragma unroll
      for (int v = 0; v < 8; ++v) {
        float r = acc[v] + red_lds[(wave & 7) * 256 + v * 32 + lane];
        __builtin_nontemporal_store(
            r, &outF[(size_t)(row0 + v + 8 * hi) * DK + n0 + ln]);
      }
    }
  }
}

// ---------------------------------------------------------------------------
extern "C" void kernel_launch(void* const* d_in, const int* in_sizes, int n_in,
                              void* d_out, int out_size, void* d_ws, size_t ws_size,
                              hipStream_t stream) {
  const float* Q   = (const float*)d_in[0];   // [16,2048,128] fp32
  const float* mem = (const float*)d_in[1];   // [4096,128]    fp32

  __bf16* memB = (__bf16*)d_ws;                       // 1 MB
  __bf16* memT = memB + (size_t)MK * DK;              // 1 MB

  float* outF = (float*)d_out;                        // memory_features first
  float* outA = outF + (size_t)BT * DK;               // then attention

  cvt_mem_kernel<<<(MK * DK) / 256, 256, 0, stream>>>(mem, memB, memT);
  fused_attn_kernel<<<BT / ROWS, THREADS, 0, stream>>>(Q, memB, memT, outF, outA);
}
